// MambaSelf_42623255446109
// MI455X (gfx1250) — matrix operand order repaired and checked by reading, not compiled, not run on hardware
//
#include <hip/hip_runtime.h>
#include <hip/hip_bf16.h>

// Problem constants (match reference)
#define HH   768
#define DMC  1536
#define P2   3072
#define NS   16
#define KC   4
#define RR   48
#define BB   2
#define LL   1024
#define BL   (BB * LL)   // 2048
#define RN   80          // R + 2N
#define RN2  160         // both directions stacked
#define KCH  128         // K chunk staged in LDS

typedef __attribute__((ext_vector_type(16))) __bf16 v16bf;
typedef __attribute__((ext_vector_type(8)))  float  v8f;

struct alignas(16) U4 { unsigned int x, y, z, w; };
union Frag { v16bf v; U4 q[2]; };

__device__ __forceinline__ unsigned short f2bf(float f) {
  unsigned int u = __float_as_uint(f);
  u += 0x7FFFu + ((u >> 16) & 1u);   // round-to-nearest-even
  return (unsigned short)(u >> 16);
}

// Async global->LDS 16B copy (per lane), tracked by ASYNCcnt.
__device__ __forceinline__ void async_copy16(unsigned int ldsAddr, const void* g) {
  asm volatile("global_load_async_to_lds_b128 %0, %1, off"
               :: "v"(ldsAddr), "v"((unsigned long long)(uintptr_t)g)
               : "memory");
}
__device__ __forceinline__ void wait_async0() {
  asm volatile("s_wait_asynccnt 0x0" ::: "memory");
}

// ---------------------------------------------------------------- casts
__global__ void cast_bf16_kernel(const float* __restrict__ src,
                                 unsigned short* __restrict__ dst, int n) {
  int i = blockIdx.x * blockDim.x + threadIdx.x;
  if (i < n) dst[i] = f2bf(src[i]);
}

// ------------------------------------------------- WMMA GEMM (bf16->f32)
// Fragment layouts (ISA 7.12.2, 16-bit):
// A frag: lane (half=lane>>4, r=lane&15) holds row r,
//   elems 0..7  = k in [half*8,  half*8+7]
//   elems 8..15 = k in [16+half*8, 16+half*8+7]
// B frag: lane holds column c=lane&15, same k layout as A frag.
// C/D:   VGPR i -> M = half*8 + i, N = lane&15.

// Small generic kernel: one 16x16 tile per wave (used for the 160-col x_proj).
__global__ void wmma_gemm_bf16(const unsigned short* __restrict__ A,
                               const unsigned short* __restrict__ W,
                               float* __restrict__ Dst,
                               int Kdim, int lda, int ldb, int ldd) {
  const int lane = threadIdx.x & 31;
  const int half = lane >> 4;
  const int lr   = lane & 15;
  const int wavesPerBlk = blockDim.x >> 5;
  const int tileN = blockIdx.x * wavesPerBlk + (threadIdx.x >> 5);
  const int tileM = blockIdx.y;

  const unsigned short* aPtr = A + (size_t)(tileM * 16 + lr) * lda;
  const unsigned short* bPtr = W + (size_t)(tileN * 16 + lr) * ldb;

  v8f acc = {};
  #pragma unroll 4
  for (int k0 = 0; k0 < Kdim; k0 += 32) {
    Frag a, b;
    a.q[0] = *(const U4*)(aPtr + k0 + half * 8);
    a.q[1] = *(const U4*)(aPtr + k0 + 16 + half * 8);
    b.q[0] = *(const U4*)(bPtr + k0 + half * 8);
    b.q[1] = *(const U4*)(bPtr + k0 + 16 + half * 8);
    acc = __builtin_amdgcn_wmma_f32_16x16x32_bf16(false, a.v, false, b.v,
                                                  (short)0, acc, false, false);
  }
  float* dBase = Dst + (size_t)(tileM * 16) * ldd + tileN * 16 + lr;
  #pragma unroll
  for (int i = 0; i < 8; ++i) {
    dBase[(size_t)(half * 8 + i) * ldd] = acc[i];
  }
}

// Big kernel: block tile 128(M)x64(N), 4 waves each doing 32x64.
// B panel (64 cols x 128 k = 16KB/chunk) is block-shared: staged in LDS
// with double-buffered GLOBAL_LOAD_ASYNC_TO_LDS_B128 (ASYNCcnt), A loaded
// directly from global. 8 WMMAs per K-step per wave.
__global__ void wmma_gemm_bf16_async(const unsigned short* __restrict__ A,
                                     const unsigned short* __restrict__ W,
                                     float* __restrict__ Dst,
                                     int Kdim, int lda, int ldb, int ldd) {
  __shared__ alignas(16) unsigned short bpanel[2][64 * KCH];   // 2 x 16KB

  const int lane = threadIdx.x & 31;
  const int half = lane >> 4;
  const int lr   = lane & 15;
  const int wave = threadIdx.x >> 5;
  const int tNb  = blockIdx.x * 64;               // N column base
  const int tM   = blockIdx.y * 8 + wave * 2;     // M tile index (16-row units)

  const unsigned short* aP[2];
  aP[0] = A + (size_t)((tM + 0) * 16 + lr) * lda;
  aP[1] = A + (size_t)((tM + 1) * 16 + lr) * lda;

  const int nc = Kdim / KCH;

  // Prologue: async-stage chunk 0 (1024 x 16B chunks, 8 per thread).
  #pragma unroll
  for (int i = 0; i < 8; ++i) {
    int idx = i * 128 + threadIdx.x;   // 0..1023
    int r   = idx >> 4;                // panel row (= W row tNb+r)
    int c16 = idx & 15;                // 16B chunk within row
    unsigned int lds = (unsigned int)(uintptr_t)&bpanel[0][r * KCH + c16 * 8];
    async_copy16(lds, W + (size_t)(tNb + r) * ldb + c16 * 8);
  }

  v8f acc[2][4] = {};
  for (int c = 0; c < nc; ++c) {
    wait_async0();        // this wave's copies for chunk c have landed
    __syncthreads();      // all waves' copies landed; prev compute finished
    if (c + 1 < nc) {
      int buf = (c + 1) & 1;
      int kb  = (c + 1) * KCH;
      #pragma unroll
      for (int i = 0; i < 8; ++i) {
        int idx = i * 128 + threadIdx.x;
        int r   = idx >> 4;
        int c16 = idx & 15;
        unsigned int lds = (unsigned int)(uintptr_t)&bpanel[buf][r * KCH + c16 * 8];
        async_copy16(lds, W + (size_t)(tNb + r) * ldb + kb + c16 * 8);
      }
    }
    const unsigned short* bp = bpanel[c & 1];
    const int kg = c * KCH;
    #pragma unroll
    for (int kk = 0; kk < KCH; kk += 32) {
      Frag a[2], b[4];
      #pragma unroll
      for (int mi = 0; mi < 2; ++mi) {
        a[mi].q[0] = *(const U4*)(aP[mi] + kg + kk + half * 8);
        a[mi].q[1] = *(const U4*)(aP[mi] + kg + kk + 16 + half * 8);
      }
      #pragma unroll
      for (int ni = 0; ni < 4; ++ni) {
        const unsigned short* bq = bp + (ni * 16 + lr) * KCH + kk;
        b[ni].q[0] = *(const U4*)(bq + half * 8);
        b[ni].q[1] = *(const U4*)(bq + 16 + half * 8);
      }
      #pragma unroll
      for (int mi = 0; mi < 2; ++mi)
        #pragma unroll
        for (int ni = 0; ni < 4; ++ni)
          acc[mi][ni] = __builtin_amdgcn_wmma_f32_16x16x32_bf16(
              false, a[mi].v, false, b[ni].v, (short)0, acc[mi][ni], false, false);
    }
  }

  #pragma unroll
  for (int mi = 0; mi < 2; ++mi) {
    #pragma unroll
    for (int ni = 0; ni < 4; ++ni) {
      float* dBase = Dst + (size_t)((tM + mi) * 16) * ldd + tNb + ni * 16 + lr;
      #pragma unroll
      for (int i = 0; i < 8; ++i)
        dBase[(size_t)(half * 8 + i) * ldd] = acc[mi][ni][i];
    }
  }
}

// --------------------------------------------- depthwise causal conv+SiLU
__global__ void conv_silu_kernel(const float* __restrict__ proj,
                                 const float* __restrict__ cw,
                                 const float* __restrict__ cb,
                                 float* __restrict__ convf,
                                 unsigned short* __restrict__ convbf) {
  int idx = blockIdx.x * blockDim.x + threadIdx.x;   // bl*DM + d
  int d  = idx % DMC;
  int bl = idx / DMC;
  int l  = bl & (LL - 1);
  int b  = bl >> 10;
  float s = cb[d];
  #pragma unroll
  for (int k = 0; k < KC; ++k) {
    int lk = l - (KC - 1) + k;
    if (lk >= 0)
      s += proj[(size_t)((b << 10) + lk) * P2 + d] * cw[d * KC + k];
  }
  float o = s / (1.0f + __expf(-s));   // SiLU
  convf[idx]  = o;
  convbf[idx] = f2bf(o);
}

// ----------------------------------------------------- delta = softplus(.)
__global__ void delta_kernel(const float* __restrict__ ssm,
                             const float* __restrict__ dtw,
                             const float* __restrict__ dtb,
                             const float* __restrict__ dtwb,
                             const float* __restrict__ dtbb,
                             float* __restrict__ deltaF,
                             float* __restrict__ deltaB) {
  __shared__ float ts[2 * RR];
  int bl = blockIdx.x;
  int t  = threadIdx.x;
  if (t < RR)           ts[t] = ssm[(size_t)bl * RN2 + t];
  else if (t < 2 * RR)  ts[t] = ssm[(size_t)bl * RN2 + RN + (t - RR)];
  __syncthreads();
  for (int d = t; d < DMC; d += blockDim.x) {
    float sf = dtb[d], sb = dtbb[d];
    const float* wf = dtw  + d * RR;
    const float* wb = dtwb + d * RR;
    #pragma unroll 8
    for (int r = 0; r < RR; ++r) {
      sf += wf[r] * ts[r];
      sb += wb[r] * ts[RR + r];
    }
    sf = (sf > 20.f) ? sf : log1pf(__expf(sf));
    sb = (sb > 20.f) ? sb : log1pf(__expf(sb));
    deltaF[(size_t)bl * DMC + d] = sf;
    deltaB[(size_t)bl * DMC + d] = sb;
  }
}

// ------------------------------------------------ selective scan (per dir)
// 16 lanes per (b,d) chain, one state n per lane; shfl_xor tree reduce.
__global__ void scan_kernel(const float* __restrict__ convf,
                            const float* __restrict__ ssm,
                            const float* __restrict__ delta,
                            const float* __restrict__ Alog,
                            const float* __restrict__ Dp,
                            float* __restrict__ ytmp,
                            int dir) {
  int grp = blockIdx.x * (blockDim.x >> 4) + (threadIdx.x >> 4);
  int n   = threadIdx.x & 15;
  int d   = grp % DMC;
  int b   = grp / DMC;
  float An = -__expf(Alog[d * NS + n]);
  float Dd = Dp[d];
  float x  = 0.f;
  int cbase = RN * dir;
  for (int t = 0; t < LL; ++t) {
    int l   = dir ? (LL - 1 - t) : t;
    int row = (b << 10) + l;
    float u  = convf[(size_t)row * DMC + d];
    float dl = delta[(size_t)row * DMC + d];
    float Bn = ssm[(size_t)row * RN2 + cbase + RR + n];
    float Cn = ssm[(size_t)row * RN2 + cbase + RR + NS + n];
    float dA = __expf(dl * An);
    x = dA * x + (dl * u) * Bn;
    float p = x * Cn;
    p += __shfl_xor(p, 1, 16);
    p += __shfl_xor(p, 2, 16);
    p += __shfl_xor(p, 4, 16);
    p += __shfl_xor(p, 8, 16);
    if (n == 0) {
      float val = p + u * Dd;
      size_t oi = (size_t)row * DMC + d;
      if (dir == 0) ytmp[oi] = val; else ytmp[oi] += val;
    }
  }
}

// -------------------------------------------- y * silu(gate) -> bf16
__global__ void finalize_kernel(const float* __restrict__ proj,
                                const float* __restrict__ ytmp,
                                unsigned short* __restrict__ ybf) {
  int idx = blockIdx.x * blockDim.x + threadIdx.x;
  int d  = idx % DMC;
  int bl = idx / DMC;
  float g = proj[(size_t)bl * P2 + DMC + d];
  float s = g / (1.0f + __expf(-g));
  ybf[idx] = f2bf(ytmp[idx] * s);
}

// =======================================================================
extern "C" void kernel_launch(void* const* d_in, const int* in_sizes, int n_in,
                              void* d_out, int out_size, void* d_ws, size_t ws_size,
                              hipStream_t stream) {
  (void)in_sizes; (void)n_in; (void)out_size; (void)ws_size;

  const float* hs    = (const float*)d_in[0];
  const float* inw   = (const float*)d_in[1];
  const float* convw = (const float*)d_in[2];
  const float* convb = (const float*)d_in[3];
  const float* xw    = (const float*)d_in[4];
  const float* dtw   = (const float*)d_in[5];
  const float* dtb   = (const float*)d_in[6];
  const float* xwb   = (const float*)d_in[7];
  const float* dtwb  = (const float*)d_in[8];
  const float* dtbb  = (const float*)d_in[9];
  const float* Alog  = (const float*)d_in[10];
  const float* Alogb = (const float*)d_in[11];
  const float* Dv    = (const float*)d_in[12];
  const float* Dvb   = (const float*)d_in[13];
  const float* ow    = (const float*)d_in[14];
  float* out = (float*)d_out;

  // workspace carve-up (256B aligned)
  char* ws = (char*)d_ws;
  size_t off = 0;
  #define WS_ALLOC(ptr_t, name, bytes) \
      ptr_t name = (ptr_t)(ws + off); off = (off + (size_t)(bytes) + 255) & ~(size_t)255;
  WS_ALLOC(float*,          proj,   (size_t)BL * P2 * 4)
  WS_ALLOC(float*,          convf,  (size_t)BL * DMC * 4)
  WS_ALLOC(float*,          deltaF, (size_t)BL * DMC * 4)
  WS_ALLOC(float*,          deltaB, (size_t)BL * DMC * 4)
  WS_ALLOC(float*,          ytmp,   (size_t)BL * DMC * 4)
  WS_ALLOC(float*,          ssm,    (size_t)BL * RN2 * 4)
  WS_ALLOC(unsigned short*, hsbf,   (size_t)BL * HH * 2)
  WS_ALLOC(unsigned short*, wbf,    (size_t)P2 * HH * 2)
  WS_ALLOC(unsigned short*, convbf, (size_t)BL * DMC * 2)
  WS_ALLOC(unsigned short*, xwbf,   (size_t)RN2 * DMC * 2)
  WS_ALLOC(unsigned short*, owbf,   (size_t)HH * DMC * 2)
  WS_ALLOC(unsigned short*, ybf,    (size_t)BL * DMC * 2)
  #undef WS_ALLOC

  // 1) bf16 casts
  {
    int n;
    n = BL * HH;   cast_bf16_kernel<<<(n + 255) / 256, 256, 0, stream>>>(hs,  hsbf, n);
    n = P2 * HH;   cast_bf16_kernel<<<(n + 255) / 256, 256, 0, stream>>>(inw, wbf,  n);
    n = RN * DMC;  cast_bf16_kernel<<<(n + 255) / 256, 256, 0, stream>>>(xw,  xwbf, n);
    n = RN * DMC;  cast_bf16_kernel<<<(n + 255) / 256, 256, 0, stream>>>(xwb, xwbf + (size_t)RN * DMC, n);
    n = HH * DMC;  cast_bf16_kernel<<<(n + 255) / 256, 256, 0, stream>>>(ow,  owbf, n);
  }

  // 2) in_proj GEMM: proj[bl][p] = sum_h hs[bl][h] * inw[p][h]
  //    block tile 128x64: grid (3072/64, 2048/128), K=768 (6 chunks)
  wmma_gemm_bf16_async<<<dim3(P2 / 64, BL / 128), 128, 0, stream>>>(
      hsbf, wbf, proj, HH, HH, HH, P2);

  // 3) causal conv + SiLU
  conv_silu_kernel<<<(BL * DMC) / 256, 256, 0, stream>>>(
      proj, convw, convb, convf, convbf);

  // 4) x_proj GEMM (both dirs stacked): ssm[bl][160], K=1536
  wmma_gemm_bf16<<<dim3(RN2 / 16 / 2, BL / 16), 64, 0, stream>>>(
      convbf, xwbf, ssm, DMC, DMC, DMC, RN2);

  // 5) delta = softplus(dt_proj), both directions
  delta_kernel<<<BL, 256, 0, stream>>>(ssm, dtw, dtb, dtwb, dtbb, deltaF, deltaB);

  // 6) selective scans (dir0 writes, dir1 accumulates)
  scan_kernel<<<(BB * DMC) / 16, 256, 0, stream>>>(convf, ssm, deltaF, Alog,  Dv,  ytmp, 0);
  scan_kernel<<<(BB * DMC) / 16, 256, 0, stream>>>(convf, ssm, deltaB, Alogb, Dvb, ytmp, 1);

  // 7) gate
  finalize_kernel<<<(BL * DMC) / 256, 256, 0, stream>>>(proj, ytmp, ybf);

  // 8) out_proj GEMM: out[bl][h] = sum_d y[bl][d] * ow[h][d]
  //    block tile 128x64: grid (768/64, 2048/128), K=1536 (12 chunks)
  wmma_gemm_bf16_async<<<dim3(HH / 64, BL / 128), 128, 0, stream>>>(
      ybf, owbf, out, DMC, DMC, DMC, HH);
}
